// CHAN_without_SA_79456894976553
// MI455X (gfx1250) — matrix operand order repaired and checked by reading, not compiled, not run on hardware
//
#include <hip/hip_runtime.h>
#include <stdint.h>

// ---------------------------------------------------------------------------
// CDNA5 (gfx1250) implementation: all heavy math on v_wmma_f32_16x16x32_bf16.
// Block tile 64(M) x 128(N), 8 waves (2x4), each wave owns 2x2 C fragments.
// Depth-2 software pipeline: regs hold steps s+1,s+2; double-buffered LDS;
// one barrier per K-step.  All global staging is b128 (uint4) per thread.
// ---------------------------------------------------------------------------

typedef __attribute__((ext_vector_type(16))) __bf16 bf16x16;
typedef __attribute__((ext_vector_type(8)))  float  f32x8;

#define DEV static __device__ __forceinline__

DEV uint16_t f32_to_bf16(float f) {
  uint32_t u = __builtin_bit_cast(uint32_t, f);
  u += 0x7FFFu + ((u >> 16) & 1u);          // round-to-nearest-even
  return (uint16_t)(u >> 16);
}
DEV float bf16_to_f32(uint16_t h) {
  uint32_t u = ((uint32_t)h) << 16;
  return __builtin_bit_cast(float, u);
}
DEV uint32_t pack_bf16x2(float lo, float hi) {
  return (uint32_t)f32_to_bf16(lo) | ((uint32_t)f32_to_bf16(hi) << 16);
}

union FragB16 { uint32_t u[8]; bf16x16 v; };

// LDS tile rows: 32 bf16 payload padded to 40 elems (80B: b128-aligned rows,
// 20-dword pitch keeps the 16 lanes of each frag half on distinct banks).
#define LDAP 40
#define ATILE (64 * LDAP)
#define BTILE (128 * LDAP)

// A-matrix 16x32 bf16 fragment (ISA 7.12.2): lane = M + 16*h,
// VGPR i<4 holds K = 8h+2i,+1 ; VGPR i>=4 holds K = 16+8h+2(i-4),+1.
DEV bf16x16 load_a_frag(const uint16_t* As, int row0) {
  const int lane = threadIdx.x & 31;
  const int m = lane & 15, h = lane >> 4;
  const uint16_t* base = As + (row0 + m) * LDAP;
  FragB16 r;
#pragma unroll
  for (int i = 0; i < 4; ++i) {
    const int k0 = 8 * h + 2 * i;
    r.u[i]     = *(const uint32_t*)(base + k0);
    r.u[i + 4] = *(const uint32_t*)(base + 16 + k0);
  }
  return r.v;
}

// B-matrix 32x16 bf16 fragment (same K order as A): VGPR i<4 holds
// K = 8h+2i,+1 ; VGPR i>=4 holds K = 16+8h+2(i-4),+1 at column n = lane&15. LDS tile stored [n][k].
DEV bf16x16 load_b_frag(const uint16_t* Bs, int col0) {
  const int lane = threadIdx.x & 31;
  const int n = lane & 15, h = lane >> 4;
  const uint16_t* base = Bs + (col0 + n) * LDAP;
  FragB16 r;
#pragma unroll
  for (int i = 0; i < 4; ++i) {
    r.u[i]     = *(const uint32_t*)(base + 8 * h + 2 * i);
    r.u[i + 4] = *(const uint32_t*)(base + 16 + 8 * h + 2 * i);
  }
  return r.v;
}

DEV f32x8 wmma_bf16(bf16x16 a, bf16x16 b, f32x8 c) {
  return __builtin_amdgcn_wmma_f32_16x16x32_bf16(false, a, false, b,
                                                 (short)0, c, false, false);
}

// ---------------------------------------------------------------------------
// Pipelined GEMM engine.
//   loadA(s)  -> this thread's uint4 (8 bf16) of the 64x32 A tile of step s
//                (thread t covers row t>>2, elems (t&3)*8 ..+7)
//   loadB(s,j)-> uint4 of the 128x32 B tile, chunk j covers row (t+j*256)>>2
//   pref(s)   -> optional global_prefetch_b8 hint
// ---------------------------------------------------------------------------
template<int KSTEPS, class FA, class FB, class FP>
DEV void gemm_pipeline(FA loadA, FB loadB, FP pref,
                       uint16_t* As, uint16_t* Bs, f32x8 (&acc)[2][2]) {
  const int tid  = threadIdx.x;
  const int wave = tid >> 5;
  const int wm = wave >> 2, wn = wave & 3;
  const int rowA = tid >> 2;
  const int cT   = (tid & 3) * 8;

  uint4 ra[2], rb[2][2];
  auto fetch = [&](int s, int slot) {
    ra[slot]    = loadA(s);
    rb[slot][0] = loadB(s, 0);
    rb[slot][1] = loadB(s, 1);
  };
  auto stage = [&](int slot, int buf) {
    *(uint4*)(As + buf * ATILE + rowA * LDAP + cT) = ra[slot];
#pragma unroll
    for (int j = 0; j < 2; ++j) {
      const int rowB = (tid + j * 256) >> 2;
      *(uint4*)(Bs + buf * BTILE + rowB * LDAP + cT) = rb[slot][j];
    }
  };

  fetch(0, 0);
  stage(0, 0);
  if (KSTEPS > 1) fetch(1, 1);
  __syncthreads();

#pragma unroll 2
  for (int s = 0; s < KSTEPS; ++s) {
    if (s + 2 < KSTEPS) fetch(s + 2, s & 1);   // ~2 wmma-stages of latency cover
    if (s + 3 < KSTEPS) pref(s + 3);           // global_prefetch_b8 hint
    const uint16_t* Ac = As + (s & 1) * ATILE;
    const uint16_t* Bc = Bs + (s & 1) * BTILE;
    const bf16x16 a0 = load_a_frag(Ac, wm * 16);
    const bf16x16 a1 = load_a_frag(Ac, wm * 16 + 32);
    const bf16x16 b0 = load_b_frag(Bc, wn * 16);
    const bf16x16 b1 = load_b_frag(Bc, wn * 16 + 64);
    acc[0][0] = wmma_bf16(a0, b0, acc[0][0]);
    acc[0][1] = wmma_bf16(a0, b1, acc[0][1]);
    acc[1][0] = wmma_bf16(a1, b0, acc[1][0]);
    acc[1][1] = wmma_bf16(a1, b1, acc[1][1]);
    if (s + 1 < KSTEPS) stage((s + 1) & 1, (s + 1) & 1);
    __syncthreads();
  }
}

// ---------------------------------------------------------------------------
// Prep kernels: dtype conversion + weight re-layout
// ---------------------------------------------------------------------------

__global__ void cvt_f32_to_bf16_kernel(const float* __restrict__ in,
                                       uint16_t* __restrict__ out,
                                       long long n) {
  long long i = (long long)blockIdx.x * blockDim.x + threadIdx.x;
  if (i < n) out[i] = f32_to_bf16(in[i]);
}

// conv weight [CO][CI][K] f32 -> [k][co][ci] bf16
__global__ void prep_conv_w_kernel(const float* __restrict__ w,
                                   uint16_t* __restrict__ o,
                                   int CO, int CI, int K) {
  long long i = (long long)blockIdx.x * blockDim.x + threadIdx.x;
  long long tot = (long long)CO * CI * K;
  if (i >= tot) return;
  int ci = (int)(i % CI);
  long long r = i / CI;
  int co = (int)(r % CO);
  int k  = (int)(r / CO);
  o[i] = f32_to_bf16(w[((size_t)co * CI + ci) * K + k]);
}

// deconv weight [CI][CO][K] f32 -> [k][co][ci] bf16
__global__ void prep_deconv_w_kernel(const float* __restrict__ w,
                                     uint16_t* __restrict__ o,
                                     int CI, int CO, int K) {
  long long i = (long long)blockIdx.x * blockDim.x + threadIdx.x;
  long long tot = (long long)CO * CI * K;
  if (i >= tot) return;
  int ci = (int)(i % CI);
  long long r = i / CI;
  int co = (int)(r % CO);
  int k  = (int)(r / CO);
  o[i] = f32_to_bf16(w[((size_t)ci * CO + co) * K + k]);
}

// q[e][b] = concept_e[b] @ w_ca1^T (256)
// u[e][b] = w_sim1^T ((concept_e[b] @ w_sim2^T) * w_mlp)  (256)
//   -- exact algebraic collapse of the sim1/MLP head:
//   (c ⊙ (d@w_sim1^T)) @ w_mlp == d @ (w_sim1^T (c ⊙ w_mlp))
__global__ __launch_bounds__(256)
void prep_vecs_kernel(const float* __restrict__ concept1,
                      const float* __restrict__ concept2,
                      const float* __restrict__ w_ca1,   // [256][300]
                      const float* __restrict__ w_sim2,  // [1024][300]
                      const float* __restrict__ w_sim1,  // [1024][256]
                      const float* __restrict__ w_mlp,   // [1024]
                      float* __restrict__ qbuf,          // [2][4][256]
                      float* __restrict__ ubuf) {        // [2][4][256]
  const int b = blockIdx.x, e = blockIdx.y;
  const float* con = (e == 0 ? concept1 : concept2) + b * 300;
  __shared__ float v_sh[1024];
  const int t = threadIdx.x;
  if (t < 256) {
    float acc = 0.f;
    const float* w = w_ca1 + t * 300;
    for (int s = 0; s < 300; ++s) acc += con[s] * w[s];
    qbuf[((size_t)e * 4 + b) * 256 + t] = acc;
  }
  for (int o = t; o < 1024; o += 256) {
    float acc = 0.f;
    const float* w = w_sim2 + (size_t)o * 300;
    for (int s = 0; s < 300; ++s) acc += con[s] * w[s];
    v_sh[o] = acc * w_mlp[o];
  }
  __syncthreads();
  if (t < 256) {
    float acc = 0.f;
    for (int s = 0; s < 1024; ++s) acc += w_sim1[(size_t)s * 256 + t] * v_sh[s];
    ubuf[((size_t)e * 4 + b) * 256 + t] = acc;
  }
}

// ---------------------------------------------------------------------------
// Conv1d(pad=2) + bias + MaxPool2, implicit GEMM on WMMA.
// AF32: A source is fp32 (raw batch) converted to bf16 during staging.
// x:[80][LIN][CIN], w:[5][COUT][CIN] bf16 -> out:[80][LIN/2][COUT] bf16
// ---------------------------------------------------------------------------
template<int LIN, int CIN, int COUT, bool AF32>
__global__ __launch_bounds__(256)
void conv_pool_kernel(const void*     __restrict__ xin,
                      const uint16_t* __restrict__ w,
                      const float*    __restrict__ bias,
                      uint16_t*       __restrict__ out) {
  constexpr int LOUT = LIN / 2;
  constexpr int KCH  = CIN / 32;
  constexpr int KSTEPS = 5 * KCH;
  const int n  = blockIdx.x;
  const int l0 = blockIdx.y * 64;
  const int c0 = blockIdx.z * 128;
  const int tid = threadIdx.x;
  const int rowT = tid >> 2, cT = (tid & 3) * 8;

  __shared__ alignas(16) uint16_t As[2 * ATILE];
  __shared__ alignas(16) uint16_t Bs[2 * BTILE];

  const float*    xf = (const float*)xin    + (size_t)n * LIN * CIN;
  const uint16_t* xh = (const uint16_t*)xin + (size_t)n * LIN * CIN;

  auto loadA = [&](int s) -> uint4 {
    const int tap = s / KCH, kk = (s % KCH) * 32;
    const int l  = l0 + rowT + tap - 2;                    // pad-2 halo
    const int lc = l < 0 ? 0 : (l >= LIN ? LIN - 1 : l);
    uint4 v;
    if constexpr (AF32) {
      const float4 f0 = *(const float4*)(xf + (size_t)lc * CIN + kk + cT);
      const float4 f1 = *(const float4*)(xf + (size_t)lc * CIN + kk + cT + 4);
      v.x = pack_bf16x2(f0.x, f0.y); v.y = pack_bf16x2(f0.z, f0.w);
      v.z = pack_bf16x2(f1.x, f1.y); v.w = pack_bf16x2(f1.z, f1.w);
    } else {
      v = *(const uint4*)(xh + (size_t)lc * CIN + kk + cT);
    }
    const uint32_t m = (l == lc) ? ~0u : 0u;               // branchless mask
    v.x &= m; v.y &= m; v.z &= m; v.w &= m;
    return v;
  };
  auto loadB = [&](int s, int j) -> uint4 {
    const int tap = s / KCH, kk = (s % KCH) * 32;
    const int rowB = (tid + j * 256) >> 2;
    return *(const uint4*)(w + ((size_t)tap * COUT + c0 + rowB) * CIN + kk + cT);
  };
  auto pref = [&](int s) {
    const int tap = s / KCH, kk = (s % KCH) * 32;
    __builtin_prefetch(w + ((size_t)tap * COUT + c0 + rowT) * CIN + kk, 0, 1);
  };

  f32x8 acc[2][2] = {};
  gemm_pipeline<KSTEPS>(loadA, loadB, pref, As, Bs, acc);

  // Fused bias + pool: C-frag VGPR r holds M=r (lanes<16) / M=8+r (lanes>=16),
  // so pooled pairs (2i,2i+1) are register-local per lane.
  const int wave = tid >> 5, lane = tid & 31;
  const int wm = wave >> 2, wn = wave & 3;
  const int h = lane >> 4;
#pragma unroll
  for (int nf = 0; nf < 2; ++nf) {
    const int co = c0 + wn * 16 + nf * 64 + (lane & 15);
    const float bv = bias[co];
#pragma unroll
    for (int mf = 0; mf < 2; ++mf) {
      const int jbase = (l0 + wm * 16 + mf * 32) / 2;
#pragma unroll
      for (int i = 0; i < 4; ++i) {
        const float p = fmaxf(acc[mf][nf][2 * i] + bv, acc[mf][nf][2 * i + 1] + bv);
        const int j = jbase + i + 4 * h;
        if (j < LOUT) out[((size_t)n * LOUT + j) * COUT + co] = f32_to_bf16(p);
      }
    }
  }
}

// ---------------------------------------------------------------------------
// kproj = t2flat[M][256] @ w_ca2^T, f32 out.  B prepped bf16 [n][k].
// ---------------------------------------------------------------------------
__global__ __launch_bounds__(256)
void gemm_k256_kernel(const uint16_t* __restrict__ A,  // [M][256]
                      const uint16_t* __restrict__ B,  // [256][256] (n-major)
                      float* __restrict__ C, int M) {
  const int m0 = blockIdx.x * 64;
  const int n0 = blockIdx.y * 128;
  const int tid = threadIdx.x;
  const int rowT = tid >> 2, cT = (tid & 3) * 8;

  __shared__ alignas(16) uint16_t As[2 * ATILE];
  __shared__ alignas(16) uint16_t Bs[2 * BTILE];

  auto loadA = [&](int s) -> uint4 {
    const int kk = s * 32;
    const int m  = m0 + rowT;
    const int mc = m >= M ? M - 1 : m;
    uint4 v = *(const uint4*)(A + (size_t)mc * 256 + kk + cT);
    const uint32_t msk = (m == mc) ? ~0u : 0u;
    v.x &= msk; v.y &= msk; v.z &= msk; v.w &= msk;
    return v;
  };
  auto loadB = [&](int s, int j) -> uint4 {
    const int kk = s * 32;
    const int rowB = (tid + j * 256) >> 2;
    return *(const uint4*)(B + (size_t)(n0 + rowB) * 256 + kk + cT);
  };
  auto pref = [&](int) {};

  f32x8 acc[2][2] = {};
  gemm_pipeline<8>(loadA, loadB, pref, As, Bs, acc);

  const int wave = tid >> 5, lane = tid & 31;
  const int wm = wave >> 2, wn = wave & 3;
  const int h = lane >> 4;
#pragma unroll
  for (int nf = 0; nf < 2; ++nf) {
    const int nn = n0 + wn * 16 + nf * 64 + (lane & 15);
#pragma unroll
    for (int mf = 0; mf < 2; ++mf) {
#pragma unroll
      for (int e = 0; e < 8; ++e) {
        const int m = m0 + wm * 16 + mf * 32 + e + 8 * h;
        if (m < M) C[(size_t)m * 256 + nn] = acc[mf][nf][e];
      }
    }
  }
}

// ---------------------------------------------------------------------------
// Additive attention per (segment n, concept e): score row is constant over
// query frames, so compute one row, masked softmax, one weighted sum.
// ---------------------------------------------------------------------------
__global__ __launch_bounds__(256)
void attn_kernel(const float* __restrict__ kproj,   // [80][50][256] f32
                 const uint16_t* __restrict__ t2,   // [80][50][256] bf16
                 const float* __restrict__ qbuf,    // [2][4][256]
                 const float* __restrict__ w_ca3,   // [256]
                 const int* __restrict__ seg_len,   // [80]
                 uint16_t* __restrict__ rbuf) {     // [80][2][256] bf16
  const int n = blockIdx.x, e = blockIdx.y;
  const int b = n / 20;
  __shared__ float a_sh[64];
  const float* q = qbuf + ((size_t)e * 4 + b) * 256;
  const int t = threadIdx.x;

  if (t < 50) {
    float acc = 0.f;
    const float* kp = kproj + ((size_t)n * 50 + t) * 256;
    for (int c = 0; c < 256; ++c) acc += tanhf(q[c] + kp[c]) * w_ca3[c];
    a_sh[t] = acc;
  }
  __syncthreads();
  if (t == 0) {
    const int lim = (seg_len[n] + 3) >> 2;   // ceil(seg_len/4), >= 1
    float mx = -3.0e38f;
    for (int j = 0; j < lim && j < 50; ++j) mx = fmaxf(mx, a_sh[j]);
    float sum = 0.f;
    for (int j = 0; j < 50; ++j) {
      float wv = (j < lim) ? expf(a_sh[j] - mx) : 0.f;
      a_sh[j] = wv;
      sum += wv;
    }
    const float inv = 1.f / sum;
    for (int j = 0; j < 50; ++j) a_sh[j] *= inv;
  }
  __syncthreads();
  if (t < 256) {
    float r = 0.f;
    for (int j = 0; j < 50; ++j)
      r += a_sh[j] * bf16_to_f32(t2[((size_t)n * 50 + j) * 256 + t]);
    rbuf[((size_t)n * 2 + e) * 256 + t] = f32_to_bf16(r);
  }
}

// ---------------------------------------------------------------------------
// ConvTranspose1d(k=4,s=2,p=1) as parity-split 2-tap GEMM:
//   y[2i]   = x[i]@W1 + x[i-1]@W3 ;  y[2i+1] = x[i+1]@W0 + x[i]@W2
// FCAT: A rows are the virtual concat {t2[n][src], r1[n], r2[n]} (no
// materialized cat buffer; the segment choice is uniform per K-chunk).
// blockIdx.z = co_tile*2 + parity
// ---------------------------------------------------------------------------
template<int LIN, int CIN, int COUT, bool FCAT>
__global__ __launch_bounds__(256)
void deconv_kernel(const uint16_t* __restrict__ x,     // [80][LIN][CIN] or t2
                   const uint16_t* __restrict__ rbuf,  // [80][2][256] (FCAT)
                   const uint16_t* __restrict__ w,     // [4][COUT][CIN] bf16
                   const float*    __restrict__ bias,  // [COUT]
                   uint16_t*       __restrict__ out) { // [80][2*LIN][COUT]
  constexpr int KCH = CIN / 32;
  constexpr int KSTEPS = 2 * KCH;
  const int n  = blockIdx.x;
  const int i0 = blockIdx.y * 64;
  const int p  = blockIdx.z & 1;
  const int c0 = (blockIdx.z >> 1) * 128;
  const int tid = threadIdx.x;
  const int rowT = tid >> 2, cT = (tid & 3) * 8;

  __shared__ alignas(16) uint16_t As[2 * ATILE];
  __shared__ alignas(16) uint16_t Bs[2 * BTILE];

  const uint16_t* xn = x + (size_t)n * LIN * (FCAT ? 256 : CIN);
  const uint16_t* r0 = FCAT ? rbuf + ((size_t)n * 2 + 0) * 256 : nullptr;
  const uint16_t* r1 = FCAT ? rbuf + ((size_t)n * 2 + 1) * 256 : nullptr;

  auto loadA = [&](int s) -> uint4 {
    const int ti = s / KCH, kk = (s % KCH) * 32;
    const int off = ti ? (p ? 0 : -1) : (p ? 1 : 0);
    const int src = i0 + rowT + off;
    const int sc  = src < 0 ? 0 : (src >= LIN ? LIN - 1 : src);
    uint4 v;
    if constexpr (FCAT) {
      const int k = kk + cT;                 // 0..767, segment uniform/chunk
      const uint16_t* ptr;
      if (k < 256)      ptr = xn + (size_t)sc * 256 + k;
      else if (k < 512) ptr = r0 + (k - 256);
      else              ptr = r1 + (k - 512);
      v = *(const uint4*)ptr;
    } else {
      v = *(const uint4*)(xn + (size_t)sc * CIN + kk + cT);
    }
    const uint32_t m = (src == sc) ? ~0u : 0u;
    v.x &= m; v.y &= m; v.z &= m; v.w &= m;
    return v;
  };
  auto loadB = [&](int s, int j) -> uint4 {
    const int ti = s / KCH, kk = (s % KCH) * 32;
    const int tap = ti ? (p ? 2 : 3) : (p ? 0 : 1);
    const int rowB = (tid + j * 256) >> 2;
    return *(const uint4*)(w + ((size_t)tap * COUT + c0 + rowB) * CIN + kk + cT);
  };
  auto pref = [&](int s) {
    const int ti = s / KCH, kk = (s % KCH) * 32;
    const int tap = ti ? (p ? 2 : 3) : (p ? 0 : 1);
    __builtin_prefetch(w + ((size_t)tap * COUT + c0 + rowT) * CIN + kk, 0, 1);
  };

  f32x8 acc[2][2] = {};
  gemm_pipeline<KSTEPS>(loadA, loadB, pref, As, Bs, acc);

  const int wave = tid >> 5, lane = tid & 31;
  const int wm = wave >> 2, wn = wave & 3;
  const int h = lane >> 4;
#pragma unroll
  for (int nf = 0; nf < 2; ++nf) {
    const int co = c0 + wn * 16 + nf * 64 + (lane & 15);
    const float bv = bias[co];
#pragma unroll
    for (int mf = 0; mf < 2; ++mf) {
#pragma unroll
      for (int e = 0; e < 8; ++e) {
        const int i = i0 + wm * 16 + mf * 32 + e + 8 * h;
        if (i < LIN) {
          const int l = 2 * i + p;
          out[((size_t)n * (2 * LIN) + l) * COUT + co] =
              f32_to_bf16(acc[mf][nf][e] + bv);
        }
      }
    }
  }
}

// ---------------------------------------------------------------------------
// Final: sc_e[row] = sigmoid(dot(d[row], u_e[b]) + b_mlp).  Wave per row.
// ---------------------------------------------------------------------------
__global__ __launch_bounds__(256)
void score_kernel(const uint16_t* __restrict__ d,   // [16000][256] bf16
                  const float* __restrict__ ubuf,   // [2][4][256]
                  const float* __restrict__ b_mlp,  // [1]
                  float* __restrict__ out) {        // [2][16000]
  const int row  = blockIdx.x * 8 + (threadIdx.x >> 5);
  const int lane = threadIdx.x & 31;
  if (row >= 16000) return;
  const int b = row / 4000;
  const uint16_t* dr = d + (size_t)row * 256;
  const float* u1 = ubuf + (size_t)b * 256;
  const float* u2 = ubuf + (size_t)(4 + b) * 256;
  float s1 = 0.f, s2 = 0.f;
#pragma unroll
  for (int c = lane; c < 256; c += 32) {
    const float v = bf16_to_f32(dr[c]);
    s1 += v * u1[c];
    s2 += v * u2[c];
  }
#pragma unroll
  for (int off = 16; off > 0; off >>= 1) {
    s1 += __shfl_down(s1, off, 32);
    s2 += __shfl_down(s2, off, 32);
  }
  if (lane == 0) {
    const float bm = b_mlp[0];
    out[row]         = 1.f / (1.f + expf(-(s1 + bm)));
    out[16000 + row] = 1.f / (1.f + expf(-(s2 + bm)));
  }
}

// ---------------------------------------------------------------------------
// Host launcher
// ---------------------------------------------------------------------------
extern "C" void kernel_launch(void* const* d_in, const int* in_sizes, int n_in,
                              void* d_out, int out_size, void* d_ws, size_t ws_size,
                              hipStream_t stream) {
  (void)in_sizes; (void)n_in; (void)out_size; (void)ws_size;

  const float* batch    = (const float*)d_in[0];
  const int*   seg_len  = (const int*)  d_in[1];
  const float* concept1 = (const float*)d_in[2];
  const float* concept2 = (const float*)d_in[3];
  const float* w_conv1  = (const float*)d_in[4];
  const float* b_conv1  = (const float*)d_in[5];
  const float* w_conv2  = (const float*)d_in[6];
  const float* b_conv2  = (const float*)d_in[7];
  const float* w_ca1    = (const float*)d_in[8];
  const float* w_ca2    = (const float*)d_in[9];
  const float* w_ca3    = (const float*)d_in[10];
  const float* w_dc1    = (const float*)d_in[11];
  const float* b_dc1    = (const float*)d_in[12];
  const float* w_dc2    = (const float*)d_in[13];
  const float* b_dc2    = (const float*)d_in[14];
  const float* w_sim1   = (const float*)d_in[15];
  const float* w_sim2   = (const float*)d_in[16];
  const float* w_mlp    = (const float*)d_in[17];
  const float* b_mlp    = (const float*)d_in[18];
  float* out = (float*)d_out;

  // workspace carve-up (256B aligned)
  char* base = (char*)d_ws;
  size_t off = 0;
  auto alloc = [&](size_t bytes) -> char* {
    char* p = base + off;
    off += (bytes + 255) & ~(size_t)255;
    return p;
  };
  uint16_t* wb1   = (uint16_t*)alloc((size_t)5 * 512 * 2048 * 2);  // [tap][co][ci]
  uint16_t* wb2   = (uint16_t*)alloc((size_t)5 * 256 * 512 * 2);
  uint16_t* wca2b = (uint16_t*)alloc((size_t)256 * 256 * 2);
  uint16_t* wd1b  = (uint16_t*)alloc((size_t)4 * 512 * 768 * 2);
  uint16_t* wd2b  = (uint16_t*)alloc((size_t)4 * 256 * 512 * 2);
  uint16_t* t1    = (uint16_t*)alloc((size_t)80 * 100 * 512 * 2);
  uint16_t* t2    = (uint16_t*)alloc((size_t)80 * 50 * 256 * 2);
  float*    kproj = (float*)   alloc((size_t)80 * 50 * 256 * 4);
  uint16_t* rbuf  = (uint16_t*)alloc((size_t)80 * 2 * 256 * 2);
  uint16_t* out1  = (uint16_t*)alloc((size_t)80 * 100 * 512 * 2);
  uint16_t* dbuf  = (uint16_t*)alloc((size_t)80 * 200 * 256 * 2);
  float*    qbuf  = (float*)   alloc((size_t)2 * 4 * 256 * 4);
  float*    ubuf  = (float*)   alloc((size_t)2 * 4 * 256 * 4);

  const int T = 256;
  auto blocks = [](long long n, int t) { return (unsigned)((n + t - 1) / t); };

  // ---- prep (weights only; batch converts inside conv1's A-loader) ----
  prep_conv_w_kernel<<<blocks(5LL * 512 * 2048, T), T, 0, stream>>>(w_conv1, wb1, 512, 2048, 5);
  prep_conv_w_kernel<<<blocks(5LL * 256 * 512, T), T, 0, stream>>>(w_conv2, wb2, 256, 512, 5);
  cvt_f32_to_bf16_kernel<<<blocks(65536LL, T), T, 0, stream>>>(w_ca2, wca2b, 65536LL);
  prep_deconv_w_kernel<<<blocks(4LL * 512 * 768, T), T, 0, stream>>>(w_dc1, wd1b, 768, 512, 4);
  prep_deconv_w_kernel<<<blocks(4LL * 256 * 512, T), T, 0, stream>>>(w_dc2, wd2b, 512, 256, 4);
  prep_vecs_kernel<<<dim3(4, 2), T, 0, stream>>>(concept1, concept2, w_ca1,
                                                 w_sim2, w_sim1, w_mlp, qbuf, ubuf);

  // ---- encoder (WMMA) ----
  conv_pool_kernel<200, 2048, 512, true><<<dim3(80, 4, 4), T, 0, stream>>>(batch, wb1, b_conv1, t1);
  conv_pool_kernel<100, 512, 256, false><<<dim3(80, 2, 2), T, 0, stream>>>(t1, wb2, b_conv2, t2);

  // ---- attention ----
  gemm_k256_kernel<<<dim3(63, 2), T, 0, stream>>>(t2, wca2b, kproj, 4000);
  attn_kernel<<<dim3(80, 2), T, 0, stream>>>(kproj, t2, qbuf, w_ca3, seg_len, rbuf);

  // ---- decoder (WMMA, cat fused into deconv1's A loader) ----
  deconv_kernel<50, 768, 512, true><<<dim3(80, 1, 8), T, 0, stream>>>(t2, rbuf, wd1b, b_dc1, out1);
  deconv_kernel<100, 512, 256, false><<<dim3(80, 2, 4), T, 0, stream>>>(out1, nullptr, wd2b, b_dc2, dbuf);

  // ---- collapsed sim/MLP head ----
  score_kernel<<<2000, T, 0, stream>>>(dbuf, ubuf, b_mlp, out);
}
